// GNN_6193342841619
// MI455X (gfx1250) — matrix operand order repaired and checked by reading, not compiled, not run on hardware
//
#include <hip/hip_runtime.h>
#include <hip/hip_bf16.h>
#include <math.h>

// ---------------- constants (match reference) ----------------
#define NHEAD   5
#define CHID    64
#define HC      320          // NHEAD*CHID = H*C_HID = H*C_OUT
#define N_CUST  50000
#define N_PROD  10000
#define N_EDGES 200000
#define N_LABEL 100000
#define EDGE_DIM 13

typedef __attribute__((ext_vector_type(16))) __bf16 v16bf;
typedef __attribute__((ext_vector_type(8)))  __bf16 v8bf;
typedef __attribute__((ext_vector_type(8)))  float  v8f;

__device__ inline v16bf cat16(v8bf x, v8bf y) {
    return __builtin_shufflevector(x, y, 0, 1, 2, 3, 4, 5, 6, 7,
                                         8, 9, 10, 11, 12, 13, 14, 15);
}

__device__ inline v8f wmma_bf16(v16bf a, v16bf b, v8f c) {
    return __builtin_amdgcn_wmma_f32_16x16x32_bf16(
        /*neg_a=*/false, a, /*neg_b=*/false, b,
        /*c_mod=*/(short)0, c, /*reuse_a=*/false, /*reuse_b=*/false);
}

__device__ inline float apply_act(float v, int act) {
    if (act == 1) return fmaxf(v, 0.0f);           // relu
    if (act == 2) return v > 0.0f ? v : 0.01f * v; // leaky 0.01 (product encoder)
    return v;
}

// ---------------------------------------------------------------------------
// Conversion pre-passes. Pad rows to Mp (mult of 32) and K to Kp (mult of 32)
// so the GEMM hot loop has zero bounds checks and zero runtime guards.
// ---------------------------------------------------------------------------
// Activations: f32 [M,K] row-major -> bf16 [Mp,Kp] row-major, zero-padded.
__global__ void aconv_kernel(const float* __restrict__ A, __bf16* __restrict__ Ab,
                             int M, int Mp, int K, int Kp) {
    long long i = (long long)blockIdx.x * blockDim.x + threadIdx.x;
    if (i >= (long long)Mp * Kp) return;
    int k = (int)(i % Kp);
    long long r = i / Kp;
    Ab[i] = (r < M && k < K) ? (__bf16)A[r * K + k] : (__bf16)0.0f;
}

// Weights: f32 [K,N] row-major -> bf16 [N,Kp] (column-major of W), zero-padded.
__global__ void wconv_kernel(const float* __restrict__ W, __bf16* __restrict__ Wt,
                             int K, int Kp, int N) {
    long long i = (long long)blockIdx.x * blockDim.x + threadIdx.x;
    if (i >= (long long)N * Kp) return;
    int k = (int)(i % Kp);
    int n = (int)(i / Kp);
    Wt[i] = (k < K) ? (__bf16)W[(long long)k * N + n] : (__bf16)0.0f;
}

// ---------------------------------------------------------------------------
// Pipelined WMMA GEMM: C[Mp,N] = act(A @ W + bias).
// A: bf16 [Mp,Kp] row-major (Mp%32==0, Kp%32==0). W: bf16 [N,Kp].
// One wave = 32 x (16*NSUB) tile; 2*NSUB WMMAs per 32-wide K chunk; the next
// chunk's fragments are loaded while the current chunk's WMMAs execute.
// No branches in the K loop; EXEC is all-ones everywhere.
// ---------------------------------------------------------------------------
template <int NSUB>
__global__ __launch_bounds__(32)
void gemm_wmma3_kernel(const __bf16* __restrict__ Ab, const __bf16* __restrict__ Wt,
                       const float* __restrict__ bias, float* __restrict__ C,
                       int Kp, int N, int act) {
    const int lane = threadIdx.x & 31;
    const int hi   = lane >> 4;
    const int lr   = lane & 15;
    const int m0   = blockIdx.y << 5;
    const int n0   = blockIdx.x * (NSUB * 16);

    const __bf16* pa0 = Ab + (long long)(m0 + lr) * Kp + hi * 8;
    const __bf16* pa1 = Ab + (long long)(m0 + 16 + lr) * Kp + hi * 8;
    const __bf16* pb[NSUB];
#pragma unroll
    for (int s = 0; s < NSUB; ++s)
        pb[s] = Wt + (long long)(n0 + s * 16 + lr) * Kp + hi * 8;

    auto ldA = [](const __bf16* p) {
        return cat16(*(const v8bf*)p, *(const v8bf*)(p + 16));
    };
    auto ldB = [](const __bf16* p) {
        return cat16(*(const v8bf*)p, *(const v8bf*)(p + 16));
    };

    // preload chunk 0
    v16bf a0c = ldA(pa0);
    v16bf a1c = ldA(pa1);
    v16bf bc[NSUB];
#pragma unroll
    for (int s = 0; s < NSUB; ++s) bc[s] = ldB(pb[s]);

    v8f acc0[NSUB] = {}, acc1[NSUB] = {};
    for (int kb = 32; kb < Kp; kb += 32) {
        v16bf a0n = ldA(pa0 + kb);
        v16bf a1n = ldA(pa1 + kb);
        v16bf bn[NSUB];
#pragma unroll
        for (int s = 0; s < NSUB; ++s) bn[s] = ldB(pb[s] + kb);
#pragma unroll
        for (int s = 0; s < NSUB; ++s) {
            acc0[s] = wmma_bf16(a0c, bc[s], acc0[s]);
            acc1[s] = wmma_bf16(a1c, bc[s], acc1[s]);
        }
        a0c = a0n;
        a1c = a1n;
#pragma unroll
        for (int s = 0; s < NSUB; ++s) bc[s] = bn[s];
    }
#pragma unroll
    for (int s = 0; s < NSUB; ++s) {
        acc0[s] = wmma_bf16(a0c, bc[s], acc0[s]);
        acc1[s] = wmma_bf16(a1c, bc[s], acc1[s]);
    }

#pragma unroll
    for (int s = 0; s < NSUB; ++s) {
        int n = n0 + s * 16 + lr;
        float bv = bias ? bias[n] : 0.0f;
#pragma unroll
        for (int r = 0; r < 8; ++r) {
            int m = m0 + r + hi * 8;
            C[(long long)m * N + n]        = apply_act(acc0[s][r] + bv, act);
            C[(long long)(m + 16) * N + n] = apply_act(acc1[s][r] + bv, act);
        }
    }
}

// ---------------------------------------------------------------------------
// Decoder layer-1: pipelined WMMA GEMM with on-the-fly gather-concat rows
// A[i] = concat(zc[ls[i]], zp[ld[i]]); K = 640, N = 64, M % 32 == 0.
// Per-group source select is wave-uniform (HC % 32 == 0) -> address cndmask.
// ---------------------------------------------------------------------------
__global__ __launch_bounds__(32)
void gemm_gather3_kernel(const __bf16* __restrict__ zcb, const __bf16* __restrict__ zpb,
                         const int* __restrict__ ls, const int* __restrict__ ld,
                         const __bf16* __restrict__ Wt, const float* __restrict__ bias,
                         float* __restrict__ C, int act) {
    const int Kp = 2 * HC;   // 640
    const int N  = 64;
    const int lane = threadIdx.x & 31;
    const int hi   = lane >> 4;
    const int lr   = lane & 15;
    const int m0   = blockIdx.y << 5;

    const int row0 = m0 + lr;
    const long long rc0 = (long long)ls[row0] * HC;
    const long long rp0 = (long long)ld[row0] * HC;
    const long long rc1 = (long long)ls[row0 + 16] * HC;
    const long long rp1 = (long long)ld[row0 + 16] * HC;

    const __bf16* pb[4];
#pragma unroll
    for (int s = 0; s < 4; ++s)
        pb[s] = Wt + (long long)(s * 16 + lr) * Kp + hi * 8;

    auto ldG = [&](long long rc, long long rp, int kb) {
        int ka = kb + hi * 8;
        int kc = kb + 16 + hi * 8;
        v8bf lo = (ka < HC) ? *(const v8bf*)(zcb + rc + ka)
                            : *(const v8bf*)(zpb + rp + (ka - HC));
        v8bf hh = (kc < HC) ? *(const v8bf*)(zcb + rc + kc)
                            : *(const v8bf*)(zpb + rp + (kc - HC));
        return cat16(lo, hh);
    };
    auto ldB = [](const __bf16* p) {
        return cat16(*(const v8bf*)p, *(const v8bf*)(p + 16));
    };

    v16bf a0c = ldG(rc0, rp0, 0);
    v16bf a1c = ldG(rc1, rp1, 0);
    v16bf bc[4];
#pragma unroll
    for (int s = 0; s < 4; ++s) bc[s] = ldB(pb[s]);

    v8f acc0[4] = {}, acc1[4] = {};
    for (int kb = 32; kb < Kp; kb += 32) {
        v16bf a0n = ldG(rc0, rp0, kb);
        v16bf a1n = ldG(rc1, rp1, kb);
        v16bf bn[4];
#pragma unroll
        for (int s = 0; s < 4; ++s) bn[s] = ldB(pb[s] + kb);
#pragma unroll
        for (int s = 0; s < 4; ++s) {
            acc0[s] = wmma_bf16(a0c, bc[s], acc0[s]);
            acc1[s] = wmma_bf16(a1c, bc[s], acc1[s]);
        }
        a0c = a0n;
        a1c = a1n;
#pragma unroll
        for (int s = 0; s < 4; ++s) bc[s] = bn[s];
    }
#pragma unroll
    for (int s = 0; s < 4; ++s) {
        acc0[s] = wmma_bf16(a0c, bc[s], acc0[s]);
        acc1[s] = wmma_bf16(a1c, bc[s], acc1[s]);
    }

#pragma unroll
    for (int s = 0; s < 4; ++s) {
        int n = s * 16 + lr;
        float bv = bias ? bias[n] : 0.0f;
#pragma unroll
        for (int r = 0; r < 8; ++r) {
            int m = m0 + r + hi * 8;
            C[(long long)m * N + n]        = apply_act(acc0[s][r] + bv, act);
            C[(long long)(m + 16) * N + n] = apply_act(acc1[s][r] + bv, act);
        }
    }
}

// ---------------------------------------------------------------------------
// GATv2 edge scoring: e[edge,h] = sum_c leaky02(xl[src]+xr[dst]+ee) * att,
// ee = edge_attr @ We computed on the fly. One wave per edge.
// ---------------------------------------------------------------------------
__global__ __launch_bounds__(256)
void edge_score_kernel(const float* __restrict__ xl, const float* __restrict__ xr,
                       const float* __restrict__ eattr, const float* __restrict__ We,
                       const float* __restrict__ att,
                       const int* __restrict__ src, const int* __restrict__ dst,
                       float* __restrict__ escore, int E) {
    int wave = (blockIdx.x * blockDim.x + threadIdx.x) >> 5;
    int lane = threadIdx.x & 31;
    if (wave >= E) return;
    const int s = src[wave], d = dst[wave];
    float ea[EDGE_DIM];
#pragma unroll
    for (int i = 0; i < EDGE_DIM; ++i) ea[i] = eattr[(long long)wave * EDGE_DIM + i];

    const long long sb = (long long)s * HC;
    const long long db = (long long)d * HC;
#pragma unroll
    for (int h = 0; h < NHEAD; ++h) {
        float acc = 0.0f;
#pragma unroll
        for (int it = 0; it < 2; ++it) {
            int col = h * CHID + lane + it * 32;
            float ee = 0.0f;
#pragma unroll
            for (int i = 0; i < EDGE_DIM; ++i) ee += ea[i] * We[i * HC + col];
            float t = xl[sb + col] + xr[db + col] + ee;
            t = t > 0.0f ? t : 0.2f * t;                 // GATv2 leaky slope 0.2
            acc += t * att[col];
        }
#pragma unroll
        for (int off = 16; off > 0; off >>= 1) acc += __shfl_down(acc, off, 32);
        if (lane == 0) escore[(long long)wave * NHEAD + h] = acc;
    }
}

__device__ inline void atomicMaxF(float* addr, float val) {
    int* ai = (int*)addr;
    int old = *ai;
    while (__int_as_float(old) < val) {
        int prev = atomicCAS(ai, old, __float_as_int(val));
        if (prev == old) break;
        old = prev;
    }
}

__global__ void fill_kernel(float* p, float v, long long n) {
    long long i = (long long)blockIdx.x * blockDim.x + threadIdx.x;
    if (i < n) p[i] = v;
}

__global__ void edge_max_kernel(const float* __restrict__ escore, const int* __restrict__ dst,
                                float* __restrict__ segmax, int E) {
    int t = blockIdx.x * blockDim.x + threadIdx.x;
    if (t >= E * NHEAD) return;
    int e = t / NHEAD, h = t - e * NHEAD;
    atomicMaxF(&segmax[(long long)dst[e] * NHEAD + h], escore[t]);
}

__global__ void edge_expsum_kernel(const float* __restrict__ escore, const float* __restrict__ segmax,
                                   const int* __restrict__ dst, float* __restrict__ ex,
                                   float* __restrict__ segsum, int E) {
    int t = blockIdx.x * blockDim.x + threadIdx.x;
    if (t >= E * NHEAD) return;
    int e = t / NHEAD, h = t - e * NHEAD;
    int d = dst[e];
    float v = expf(escore[t] - segmax[(long long)d * NHEAD + h]);
    ex[t] = v;
    atomicAdd(&segsum[(long long)d * NHEAD + h], v);
}

// out[dst, :] += alpha[h] * xl[src, :]   (one wave per edge, 320 columns)
__global__ __launch_bounds__(256)
void edge_aggregate_kernel(const float* __restrict__ xl, const float* __restrict__ ex,
                           const float* __restrict__ segsum,
                           const int* __restrict__ src, const int* __restrict__ dst,
                           float* __restrict__ out, int E) {
    int wave = (blockIdx.x * blockDim.x + threadIdx.x) >> 5;
    int lane = threadIdx.x & 31;
    if (wave >= E) return;
    const int s = src[wave], d = dst[wave];
    float alpha[NHEAD];
#pragma unroll
    for (int h = 0; h < NHEAD; ++h)
        alpha[h] = ex[(long long)wave * NHEAD + h] /
                   (segsum[(long long)d * NHEAD + h] + 1e-16f);
    const long long sb = (long long)s * HC;
    const long long db = (long long)d * HC;
#pragma unroll
    for (int it = 0; it < HC / 32; ++it) {
        int col = lane + it * 32;
        int h = col >> 6;
        atomicAdd(&out[db + col], alpha[h] * xl[sb + col]);
    }
}

__global__ void bias_act_kernel(float* __restrict__ X, const float* __restrict__ bias,
                                int M, int N, int act) {
    long long i = (long long)blockIdx.x * blockDim.x + threadIdx.x;
    if (i >= (long long)M * N) return;
    X[i] = apply_act(X[i] + bias[(int)(i % N)], act);
}

// final decoder layer: out = sigmoid(h2 @ W3(32x1) + b3)
__global__ void dec_final_kernel(const float* __restrict__ h2, const float* __restrict__ W3,
                                 const float* __restrict__ b3, float* __restrict__ out, int M) {
    int i = blockIdx.x * blockDim.x + threadIdx.x;
    if (i >= M) return;
    float acc = b3[0];
#pragma unroll
    for (int j = 0; j < 32; ++j) acc += h2[(long long)i * 32 + j] * W3[j];
    out[i] = 1.0f / (1.0f + expf(-acc));
}

// ---------------------------------------------------------------------------
// Host orchestration
// ---------------------------------------------------------------------------
struct GatP { const float *We, *Wl, *Wr, *att, *b, *bl, *br; };

static inline int pad32(int x) { return (x + 31) & ~31; }

extern "C" void kernel_launch(void* const* d_in, const int* in_sizes, int n_in,
                              void* d_out, int out_size, void* d_ws, size_t ws_size,
                              hipStream_t stream) {
    // --- inputs (setup_inputs order; params pytree leaves, dict keys sorted) ---
    const float* products   = (const float*)d_in[0];   // [N_PROD, 769]
    const float* x_customer = (const float*)d_in[1];   // [N_CUST, 15]
    const float* edge_attr  = (const float*)d_in[2];   // [N_EDGES, 13]
    const float* dW1 = (const float*)d_in[3];
    const float* dW2 = (const float*)d_in[4];
    const float* dW3 = (const float*)d_in[5];
    const float* db1 = (const float*)d_in[6];
    const float* db2 = (const float*)d_in[7];
    const float* db3 = (const float*)d_in[8];
    auto gat = [&](int base) {
        GatP g;
        g.We  = (const float*)d_in[base + 0];
        g.Wl  = (const float*)d_in[base + 1];
        g.Wr  = (const float*)d_in[base + 2];
        g.att = (const float*)d_in[base + 3];
        g.b   = (const float*)d_in[base + 4];
        g.bl  = (const float*)d_in[base + 5];
        g.br  = (const float*)d_in[base + 6];
        return g;
    };
    GatP l1_c2p = gat(9),  l1_p2c = gat(16);
    GatP l2_c2p = gat(23), l2_p2c = gat(30);
    GatP l3_c2p = gat(37), l3_p2c = gat(44);
    const float* pW1 = (const float*)d_in[51];
    const float* pW2 = (const float*)d_in[52];
    const float* pW3 = (const float*)d_in[53];
    const float* pb1 = (const float*)d_in[54];
    const float* pb2 = (const float*)d_in[55];
    const float* pb3 = (const float*)d_in[56];
    const int* edge_src  = (const int*)d_in[57];
    const int* edge_dst  = (const int*)d_in[58];
    const int* label_src = (const int*)d_in[59];
    const int* label_dst = (const int*)d_in[60];

    const int MC = pad32(N_CUST);   // 50016
    const int MP = pad32(N_PROD);   // 10016
    const int ML = pad32(N_LABEL);  // 100000

    // --- workspace carve-up (f32 region then bf16 region); rows padded to 32 ---
    float* ws = (float*)d_ws;
    size_t off = 0;
    auto alloc = [&](size_t nf) { float* p = ws + off; off += nf; return p; };
    float* pe_h1  = alloc((size_t)MP * 512);
    float* pe_h2  = alloc((size_t)MP * 256);
    float* xp0    = alloc((size_t)MP * 128);
    float* xcA    = alloc((size_t)MC * HC);
    float* xcB    = alloc((size_t)MC * HC);
    float* xpA    = alloc((size_t)MP * HC);
    float* xpB    = alloc((size_t)MP * HC);
    float* xl     = alloc((size_t)MC * HC);
    float* xr     = alloc((size_t)MC * HC);
    float* escore = alloc((size_t)N_EDGES * NHEAD);
    float* ex     = alloc((size_t)N_EDGES * NHEAD);
    float* segmax = alloc((size_t)N_CUST * NHEAD);
    float* segsum = alloc((size_t)N_CUST * NHEAD);
    float* dec_h1 = alloc((size_t)ML * 64);
    float* dec_h2 = alloc((size_t)ML * 32);
    __bf16* bws = (__bf16*)(ws + off);
    size_t boff = 0;
    auto balloc = [&](size_t nb) { __bf16* p = bws + boff; boff += nb; return p; };
    __bf16* Ab  = balloc((size_t)MC * HC);     // shared A scratch (max ~16M)
    __bf16* Wtb = balloc((size_t)512 * 800);   // shared W scratch (max)
    __bf16* zcb = balloc((size_t)N_CUST * HC); // decoder zc (bf16)
    __bf16* zpb = balloc((size_t)N_PROD * HC); // decoder zp (bf16)
    (void)ws_size; (void)in_sizes; (void)n_in; (void)out_size;

    const int E = N_EDGES;

    // full GEMM: convert A (pad M,K) and W (transpose+pad K), then WMMA
    auto gemm = [&](const float* A, const float* W, const float* bias, float* C,
                    int M, int K, int N, int act) {
        int Kp = (K + 31) & ~31;
        int Mp = pad32(M);
        long long na = (long long)Mp * Kp;
        aconv_kernel<<<(unsigned)((na + 255) / 256), 256, 0, stream>>>(A, Ab, M, Mp, K, Kp);
        long long nw = (long long)N * Kp;
        wconv_kernel<<<(unsigned)((nw + 255) / 256), 256, 0, stream>>>(W, Wtb, K, Kp, N);
        if ((N & 63) == 0) {
            dim3 grid(N / 64, Mp / 32);
            gemm_wmma3_kernel<4><<<grid, 32, 0, stream>>>(Ab, Wtb, bias, C, Kp, N, act);
        } else {  // N == 32
            dim3 grid(N / 32, Mp / 32);
            gemm_wmma3_kernel<2><<<grid, 32, 0, stream>>>(Ab, Wtb, bias, C, Kp, N, act);
        }
    };

    // --- product encoder: leaky(0.01) MLP 769 -> 512 -> 256 -> 128 ---
    gemm(products, pW1, pb1, pe_h1, N_PROD, 769, 512, 2);
    gemm(pe_h1,    pW2, pb2, pe_h2, MP,     512, 256, 2);   // padded rows flow through
    gemm(pe_h2,    pW3, pb3, xp0,   MP,     256, 128, 0);

    // --- one GATv2 direction ---
    auto run_gat = [&](const float* xs, int ns, int ds,
                       const float* xd, int nd, int dd,
                       const GatP& P, const int* sidx, const int* didx,
                       float* out, int act) {
        gemm(xs, P.Wl, P.bl, xl, ns, ds, HC, 0);
        gemm(xd, P.Wr, P.br, xr, nd, dd, HC, 0);
        edge_score_kernel<<<(E + 7) / 8, 256, 0, stream>>>(
            xl, xr, edge_attr, P.We, P.att, sidx, didx, escore, E);
        long long ndh = (long long)nd * NHEAD;
        fill_kernel<<<(unsigned)((ndh + 255) / 256), 256, 0, stream>>>(segmax, -3.0e38f, ndh);
        edge_max_kernel<<<(E * NHEAD + 255) / 256, 256, 0, stream>>>(escore, didx, segmax, E);
        fill_kernel<<<(unsigned)((ndh + 255) / 256), 256, 0, stream>>>(segsum, 0.0f, ndh);
        edge_expsum_kernel<<<(E * NHEAD + 255) / 256, 256, 0, stream>>>(
            escore, segmax, didx, ex, segsum, E);
        long long ndc = (long long)nd * HC;
        fill_kernel<<<(unsigned)((ndc + 255) / 256), 256, 0, stream>>>(out, 0.0f, ndc);
        edge_aggregate_kernel<<<(E + 7) / 8, 256, 0, stream>>>(
            xl, ex, segsum, sidx, didx, out, E);
        bias_act_kernel<<<(unsigned)((ndc + 255) / 256), 256, 0, stream>>>(
            out, P.b, nd, HC, act);
    };

    // --- hetero layer 1 (relu) ---
    run_gat(x_customer, N_CUST, 15,  xp0, N_PROD, 128, l1_c2p, edge_src, edge_dst, xpA, 1);
    run_gat(xp0,        N_PROD, 128, x_customer, N_CUST, 15, l1_p2c, edge_dst, edge_src, xcA, 1);
    // --- hetero layer 2 (relu) ---
    run_gat(xcA, N_CUST, HC, xpA, N_PROD, HC, l2_c2p, edge_src, edge_dst, xpB, 1);
    run_gat(xpA, N_PROD, HC, xcA, N_CUST, HC, l2_p2c, edge_dst, edge_src, xcB, 1);
    // --- hetero layer 3 (no relu): zp = xpA, zc = xcA ---
    run_gat(xcB, N_CUST, HC, xpB, N_PROD, HC, l3_c2p, edge_src, edge_dst, xpA, 0);
    run_gat(xpB, N_PROD, HC, xcB, N_CUST, HC, l3_p2c, edge_dst, edge_src, xcA, 0);

    // --- decoder ---
    {
        long long nzc = (long long)N_CUST * HC;
        long long nzp = (long long)N_PROD * HC;
        aconv_kernel<<<(unsigned)((nzc + 255) / 256), 256, 0, stream>>>(xcA, zcb, N_CUST, N_CUST, HC, HC);
        aconv_kernel<<<(unsigned)((nzp + 255) / 256), 256, 0, stream>>>(xpA, zpb, N_PROD, N_PROD, HC, HC);
        long long nw1 = (long long)64 * 640;
        wconv_kernel<<<(unsigned)((nw1 + 255) / 256), 256, 0, stream>>>(dW1, Wtb, 640, 640, 64);
        dim3 g1(1, ML / 32);
        gemm_gather3_kernel<<<g1, 32, 0, stream>>>(
            zcb, zpb, label_src, label_dst, Wtb, db1, dec_h1, 1);
        gemm(dec_h1, dW2, db2, dec_h2, N_LABEL, 64, 32, 1);
        dec_final_kernel<<<(N_LABEL + 255) / 256, 256, 0, stream>>>(
            dec_h2, dW3, db3, (float*)d_out, N_LABEL);
    }
}
